// ClassicalSelfAttention_65481071395916
// MI455X (gfx1250) — matrix operand order repaired and checked by reading, not compiled, not run on hardware
//
#include <hip/hip_runtime.h>
#include <hip/hip_bf16.h>

typedef __bf16 bf16;
typedef __attribute__((ext_vector_type(16))) __bf16 v16bf;
typedef __attribute__((ext_vector_type(8)))  __bf16 v8bf;
typedef __attribute__((ext_vector_type(8)))  float  v8f;

#define BM 128
#define BN 128
#define BKK 32

// --- CDNA5 async global->LDS path (ASYNCcnt), with safe fallback ------------
#if defined(__gfx1250__) && __has_builtin(__builtin_amdgcn_global_load_async_to_lds_b128)
#define USE_ASYNC_LDS 1
#define AS1 __attribute__((address_space(1)))
#define AS3 __attribute__((address_space(3)))
typedef int v4i __attribute__((vector_size(16)));   // matches builtin pointee type
#else
#define USE_ASYNC_LDS 0
#endif

__device__ __forceinline__ void copy16B_to_lds(const bf16* g, bf16* l) {
#if USE_ASYNC_LDS
    // global_load_async_to_lds_b128: per-lane 16B DMA, tracked with ASYNCcnt
    __builtin_amdgcn_global_load_async_to_lds_b128((AS1 v4i*)g, (AS3 v4i*)l, 0, 0);
#else
    *(v8bf*)l = *(const v8bf*)g;
#endif
}

__device__ __forceinline__ void wait_async_fill() {
#if USE_ASYNC_LDS
#if __has_builtin(__builtin_amdgcn_s_wait_asynccnt)
    __builtin_amdgcn_s_wait_asynccnt(0);
#else
    asm volatile("s_wait_asynccnt 0" ::: "memory");
#endif
#endif
}

// ---------------------------------------------------------------------------
// fp32 -> bf16 downcast
// ---------------------------------------------------------------------------
__global__ __launch_bounds__(256)
void cvt_f32_bf16(const float* __restrict__ src, bf16* __restrict__ dst, long n) {
    long i = (long)blockIdx.x * 256 + threadIdx.x;
    if (i < n) dst[i] = (bf16)src[i];
}

// ---------------------------------------------------------------------------
// bf16 WMMA GEMM:  C[M,N] = scale * A[M,K] @ B^T   (B stored row-major [N][K])
//   outMode: 0 = f32 row-major [M][N] (ldc)
//            1 = bf16 row-major [M][N] (ldc)
//            2 = bf16 transposed [N][M] store (index col*ldc + row)
//   batched via blockIdx.z with element strides sA/sB/sC.
// Block tile 128x128x32, 256 threads = 8 waves, wave tile 32x64 (2x4 WMMA).
// LDS double-buffered; tiles staged with async global->LDS DMA.
// ---------------------------------------------------------------------------
__global__ __launch_bounds__(256)
void gemm_bf16_wmma(const bf16* __restrict__ A, const bf16* __restrict__ B,
                    void* __restrict__ Cout,
                    int K, int lda, int ldb, int ldc,
                    long sA, long sB, long sC,
                    float scale, int outMode) {
    __shared__ bf16 Asm[2][BM][BKK];   // [buf][row][k]
    __shared__ bf16 Bsm[2][BN][BKK];   // [buf][col][k] (k contiguous per column)

    const int tid   = threadIdx.x;
    const int lane  = tid & 31;
    const int wave  = tid >> 5;
    const int wm    = wave & 3;      // 4 row-strips of 32
    const int wn    = wave >> 2;     // 2 col-strips of 64
    const int khalf = lane >> 4;     // wave32 K-half select (ISA 7.12.2)
    const int l16   = lane & 15;

    const long bm0 = (long)blockIdx.x * BM;
    const long bn0 = (long)blockIdx.y * BN;
    const bf16* Ab = A + (long)blockIdx.z * sA;
    const bf16* Bb = B + (long)blockIdx.z * sB;

    // Fixed per-thread fill coordinates: 512 16-byte chunks per tile,
    // 2 chunks per thread, 4 chunks per row/col.
    const int fr0 = tid >> 2;            // rows/cols 0..63
    const int fr1 = fr0 + 64;            // rows/cols 64..127
    const int fc  = (tid & 3) * 8;       // k-offset within tile

    v8f acc[2][4];
#pragma unroll
    for (int i = 0; i < 2; ++i)
#pragma unroll
        for (int j = 0; j < 4; ++j)
#pragma unroll
            for (int r = 0; r < 8; ++r)
                acc[i][j][r] = 0.0f;

    // stage first K-tile
    copy16B_to_lds(Ab + (bm0 + fr0) * (long)lda + fc, &Asm[0][fr0][fc]);
    copy16B_to_lds(Ab + (bm0 + fr1) * (long)lda + fc, &Asm[0][fr1][fc]);
    copy16B_to_lds(Bb + (bn0 + fr0) * (long)ldb + fc, &Bsm[0][fr0][fc]);
    copy16B_to_lds(Bb + (bn0 + fr1) * (long)ldb + fc, &Bsm[0][fr1][fc]);

    int cur = 0;
    for (int k0 = 0; k0 < K; k0 += BKK) {
        wait_async_fill();        // this wave's DMA done (ASYNCcnt==0)
        __syncthreads();          // tiles for 'cur' visible to all waves

        // DMA next K-tile into the other buffer while we compute
        if (k0 + BKK < K) {
            const int nxt = cur ^ 1;
            const int kn  = k0 + BKK;
            copy16B_to_lds(Ab + (bm0 + fr0) * (long)lda + kn + fc, &Asm[nxt][fr0][fc]);
            copy16B_to_lds(Ab + (bm0 + fr1) * (long)lda + kn + fc, &Asm[nxt][fr1][fc]);
            copy16B_to_lds(Bb + (bn0 + fr0) * (long)ldb + kn + fc, &Bsm[nxt][fr0][fc]);
            copy16B_to_lds(Bb + (bn0 + fr1) * (long)ldb + kn + fc, &Bsm[nxt][fr1][fc]);
        }

        // ---- build wave32 WMMA fragments from LDS (ds_load_b128 pairs) ----
        union Frag { v16bf v; v8bf h[2]; };
        Frag a[2], b[4];
#pragma unroll
        for (int i = 0; i < 2; ++i) {
            const int row = wm * 32 + i * 16 + l16;          // A: lane = M row
            a[i].h[0] = *(const v8bf*)&Asm[cur][row][khalf * 8];       // K 0-7 / 8-15
            a[i].h[1] = *(const v8bf*)&Asm[cur][row][16 + khalf * 8];  // K 16-23 / 24-31
        }
#pragma unroll
        for (int j = 0; j < 4; ++j) {
            const int col = wn * 64 + j * 16 + l16;          // B: lane = N col
            b[j].h[0] = *(const v8bf*)&Bsm[cur][col][khalf * 8];       // K 0-7 / 8-15
            b[j].h[1] = *(const v8bf*)&Bsm[cur][col][16 + khalf * 8];  // K 16-23 / 24-31
        }

        // ---- 8 x V_WMMA_F32_16X16X32_BF16 per K-step ----
#pragma unroll
        for (int i = 0; i < 2; ++i)
#pragma unroll
            for (int j = 0; j < 4; ++j)
                acc[i][j] = __builtin_amdgcn_wmma_f32_16x16x32_bf16(
                    false, a[i].v, false, b[j].v, (short)0, acc[i][j],
                    false, false);

        __syncthreads();          // all waves finished reading 'cur'
        cur ^= 1;
    }

    // ---- epilogue: C/D layout row = r + 8*(lane>>4), col = lane&15 ----
    const long cBase = (long)blockIdx.z * sC;
#pragma unroll
    for (int i = 0; i < 2; ++i) {
#pragma unroll
        for (int j = 0; j < 4; ++j) {
            const long col = bn0 + wn * 64 + j * 16 + l16;
#pragma unroll
            for (int r = 0; r < 8; ++r) {
                const long row = bm0 + wm * 32 + i * 16 + r + 8 * khalf;
                const float v = acc[i][j][r] * scale;
                if (outMode == 0)
                    ((float*)Cout)[cBase + row * (long)ldc + col] = v;
                else if (outMode == 1)
                    ((bf16*)Cout)[cBase + row * (long)ldc + col] = (bf16)v;
                else
                    ((bf16*)Cout)[cBase + col * (long)ldc + row] = (bf16)v;
            }
        }
    }
}

// ---------------------------------------------------------------------------
// In-place row softmax on bf16 scores, f32 math. One block per row, S = 2048.
// ---------------------------------------------------------------------------
__global__ __launch_bounds__(256)
void softmax_inplace_bf16(bf16* __restrict__ P, int S) {
    const long row = blockIdx.x;
    bf16* p = P + row * (long)S;
    const int tid = threadIdx.x;

    float x[8];
    float m = -3.4e38f;
#pragma unroll
    for (int i = 0; i < 8; ++i) {
        x[i] = (float)p[tid + i * 256];
        m = fmaxf(m, x[i]);
    }
    __shared__ float red[8];
#pragma unroll
    for (int off = 16; off > 0; off >>= 1)
        m = fmaxf(m, __shfl_xor(m, off, 32));
    if ((tid & 31) == 0) red[tid >> 5] = m;
    __syncthreads();
    m = red[0];
#pragma unroll
    for (int w = 1; w < 8; ++w) m = fmaxf(m, red[w]);
    __syncthreads();

    float s = 0.0f;
#pragma unroll
    for (int i = 0; i < 8; ++i) { x[i] = __expf(x[i] - m); s += x[i]; }
#pragma unroll
    for (int off = 16; off > 0; off >>= 1)
        s += __shfl_xor(s, off, 32);
    if ((tid & 31) == 0) red[tid >> 5] = s;
    __syncthreads();
    s = 0.0f;
#pragma unroll
    for (int w = 0; w < 8; ++w) s += red[w];
    const float inv = 1.0f / s;
#pragma unroll
    for (int i = 0; i < 8; ++i)
        p[tid + i * 256] = (bf16)(x[i] * inv);
}

// ---------------------------------------------------------------------------
extern "C" void kernel_launch(void* const* d_in, const int* in_sizes, int n_in,
                              void* d_out, int out_size, void* d_ws, size_t ws_size,
                              hipStream_t stream) {
    (void)in_sizes; (void)n_in; (void)out_size; (void)ws_size;

    const float* x  = (const float*)d_in[0];
    const float* Wq = (const float*)d_in[1];
    const float* Wk = (const float*)d_in[2];
    const float* Wv = (const float*)d_in[3];
    float* out = (float*)d_out;

    const int  Bn = 4, S = 2048, D = 1024;
    const long BS = (long)Bn * S;            // 8192 rows
    const long nX = BS * D;                  // 8,388,608 elems
    const long nW = (long)D * D;

    // workspace layout (bf16 elements), ~102 MB total
    bf16* xb  = (bf16*)d_ws;
    bf16* Wqb = xb  + nX;
    bf16* Wkb = Wqb + nW;
    bf16* Wvb = Wkb + nW;
    bf16* Qb  = Wvb + nW;
    bf16* Kb  = Qb  + nX;
    bf16* Vt  = Kb  + nX;                    // V transposed: [B][e][s]
    bf16* P   = Vt  + nX;                    // Bn * S * S

    // 1) downcast activations and weights to bf16
    cvt_f32_bf16<<<dim3((unsigned)((nX + 255) / 256)), 256, 0, stream>>>(x,  xb,  nX);
    cvt_f32_bf16<<<dim3((unsigned)((nW + 255) / 256)), 256, 0, stream>>>(Wq, Wqb, nW);
    cvt_f32_bf16<<<dim3((unsigned)((nW + 255) / 256)), 256, 0, stream>>>(Wk, Wkb, nW);
    cvt_f32_bf16<<<dim3((unsigned)((nW + 255) / 256)), 256, 0, stream>>>(Wv, Wvb, nW);

    // 2) projections: Q/K = x @ W^T (bf16 row-major); V stored transposed
    dim3 gp(BS / BM, D / BN, 1);             // 64 x 8
    gemm_bf16_wmma<<<gp, 256, 0, stream>>>(xb, Wqb, Qb, D, D, D, D,
                                           0L, 0L, 0L, 1.0f, /*bf16*/1);
    gemm_bf16_wmma<<<gp, 256, 0, stream>>>(xb, Wkb, Kb, D, D, D, D,
                                           0L, 0L, 0L, 1.0f, 1);
    dim3 gv(S / BM, D / BN, Bn);             // 16 x 8 x 4 (per-batch, transposed out)
    gemm_bf16_wmma<<<gv, 256, 0, stream>>>(xb, Wvb, Vt, D, D, D, /*ldc*/S,
                                           (long)S * D, 0L, (long)D * S,
                                           1.0f, /*bf16 transposed*/2);

    // 3) scores: P = (Q @ K^T) / sqrt(D), batched over B
    dim3 gs(S / BM, S / BN, Bn);             // 16 x 16 x 4
    gemm_bf16_wmma<<<gs, 256, 0, stream>>>(Qb, Kb, P, D, D, D, S,
                                           (long)S * D, (long)S * D, (long)S * S,
                                           0.03125f /*1/sqrt(1024)*/, 1);

    // 4) softmax over rows (in place)
    softmax_inplace_bf16<<<dim3((unsigned)(Bn * (long)S)), 256, 0, stream>>>(P, S);

    // 5) out = P @ V = P @ (V^T)^T  (f32 output)
    dim3 go(S / BM, D / BN, Bn);             // 16 x 8 x 4
    gemm_bf16_wmma<<<go, 256, 0, stream>>>(P, Vt, out, S, S, S, D,
                                           (long)S * S, (long)D * S, (long)S * D,
                                           1.0f, /*f32*/0);
}
